// AttentionBlock_32427003085541
// MI455X (gfx1250) — matrix operand order repaired and checked by reading, not compiled, not run on hardware
//
#include <hip/hip_runtime.h>
#include <hip/hip_bf16.h>

typedef __attribute__((ext_vector_type(16))) _Float16 v16h;
typedef __attribute__((ext_vector_type(8)))  _Float16 h8;
typedef __attribute__((ext_vector_type(4)))  _Float16 h4;
typedef __attribute__((ext_vector_type(8)))  float    v8f;
typedef __attribute__((ext_vector_type(4)))  float    f4;

#define B_SZ   2
#define D_SZ   512
#define N_SZ   2048
#define H_SZ   8
#define HD     64     // head dim (att & out)
#define CH     512    // ATT_DIM == OUT_DIM

// ---------------------------------------------------------------------
// CDNA5 WMMA fragment loaders (ISA 7.12.2), fully vectorized:
//  A (16x32 MxK, row-major tile, pitch % 8 == 0):
//    lane(lo,hi): elements = row[hi*8..+7] ++ row[16+hi*8..+7]  (2x b128)
//  B (32x16 KxN) from a K-MAJOR (transposed) tile trow = &T[n][k0]:
//    lane(lo,hi): elements = trow[hi*8..+7] ++ trow[16+hi*8..+7] (2x b128)
// ---------------------------------------------------------------------
__device__ __forceinline__ v16h cat8(h8 a, h8 b) {
  return __builtin_shufflevector(a, b, 0, 1, 2, 3, 4, 5, 6, 7,
                                 8, 9, 10, 11, 12, 13, 14, 15);
}
__device__ __forceinline__ v16h load_a(const _Float16* row, int hi) {
  h8 c0 = *(const h8*)(row + hi * 8);
  h8 c1 = *(const h8*)(row + 16 + hi * 8);
  return cat8(c0, c1);
}
__device__ __forceinline__ v16h load_bT(const _Float16* trow, int hi) {
  h8 c0 = *(const h8*)(trow + hi * 8);
  h8 c1 = *(const h8*)(trow + 16 + hi * 8);
  return cat8(c0, c1);
}
__device__ __forceinline__ v8f wmma16(v16h a, v16h b, v8f c) {
  return __builtin_amdgcn_wmma_f32_16x16x32_f16(false, a, false, b, (short)0, c,
                                                false, false);
}
// CDNA5 async global->LDS copy (16B per lane, ASYNCcnt-tracked)
__device__ __forceinline__ void async_b128(void* lds, const void* gmem) {
  unsigned l = (unsigned)(uintptr_t)lds;
  unsigned long long g = (unsigned long long)(uintptr_t)gmem;
  asm volatile("global_load_async_to_lds_b128 %0, %1, off" ::"v"(l), "v"(g)
               : "memory");
}
__device__ __forceinline__ void wait_async0() {
  asm volatile("s_wait_asynccnt 0x0" ::: "memory");
}

// =====================================================================
// 1) Projection: y = W x + bias, output f16.
//    dlayout==0: dst[b][h][n][d]  (Q, K)      c = d*H + h
//    dlayout==1: dst[b][h][d][n]  (V, d-major for attention staging)
// =====================================================================
__global__ __launch_bounds__(128) void proj_kernel(
    const float* __restrict__ W, const float* __restrict__ bias,
    const float* __restrict__ X, _Float16* __restrict__ dst, int dlayout) {
  const int ct = blockIdx.x;            // channel tile (8)
  const int nt = blockIdx.y;            // column tile (32)
  const int b  = blockIdx.z;
  const int tid = threadIdx.x;
  const int lid = tid & 31, w = tid >> 5;
  const int lo = lid & 15, hi = lid >> 4;

  __shared__ __align__(16) _Float16 Wt[64][40];    // [c][k] row-major
  __shared__ __align__(16) _Float16 XtT[64][40];   // [n][k] k-major (transposed)

  v8f acc[4];
#pragma unroll
  for (int j = 0; j < 4; ++j)
#pragma unroll
    for (int e = 0; e < 8; ++e) acc[j][e] = 0.f;

  for (int kk = 0; kk < D_SZ; kk += 32) {
    __syncthreads();
#pragma unroll
    for (int it = 0; it < 4; ++it) {   // Wt: f32x4 load -> cvt -> h4 store
      int s = it * 128 + tid;          // 0..511
      int r = s >> 3, g = s & 7;       // row, 4-col group
      f4 x = *(const f4*)&W[(size_t)(ct * 64 + r) * D_SZ + kk + g * 4];
      *(h4*)&Wt[r][g * 4] = __builtin_convertvector(x, h4);
    }
#pragma unroll
    for (int it = 0; it < 4; ++it) {   // XtT: f32x4 along n, scatter-transpose
      int s = it * 128 + tid;          // 0..511
      int r = s >> 4;                  // k row 0..31
      int c4 = (s & 15) * 4;           // n group
      f4 x = *(const f4*)&X[((size_t)b * D_SZ + kk + r) * N_SZ + nt * 64 + c4];
      XtT[c4 + 0][r] = (_Float16)x[0];
      XtT[c4 + 1][r] = (_Float16)x[1];
      XtT[c4 + 2][r] = (_Float16)x[2];
      XtT[c4 + 3][r] = (_Float16)x[3];
    }
    __syncthreads();

    v16h a = load_a(&Wt[w * 16 + lo][0], hi);
    v16h b0 = load_bT(&XtT[0 * 16 + lo][0], hi);
    v16h b1 = load_bT(&XtT[1 * 16 + lo][0], hi);
    v16h b2 = load_bT(&XtT[2 * 16 + lo][0], hi);
    v16h b3 = load_bT(&XtT[3 * 16 + lo][0], hi);
    acc[0] = wmma16(a, b0, acc[0]);
    acc[1] = wmma16(a, b1, acc[1]);
    acc[2] = wmma16(a, b2, acc[2]);
    acc[3] = wmma16(a, b3, acc[3]);
  }

  float bia[8];
#pragma unroll
  for (int e = 0; e < 8; ++e) bia[e] = bias[ct * 64 + w * 16 + e + 8 * hi];

#pragma unroll
  for (int j = 0; j < 4; ++j) {
#pragma unroll
    for (int e = 0; e < 8; ++e) {
      int c = ct * 64 + w * 16 + e + 8 * hi;
      int n = nt * 64 + j * 16 + lo;
      int h = c & (H_SZ - 1);
      int d = c >> 3;
      size_t idx = dlayout
                       ? (((size_t)b * H_SZ + h) * HD + d) * N_SZ + n
                       : (((size_t)b * H_SZ + h) * N_SZ + n) * HD + d;
      dst[idx] = (_Float16)(acc[j][e] + bia[e]);
    }
  }
}

// =====================================================================
// 2) Fused flash attention per (64 query rows, head, batch).
//    qf,kf: [b][h][n][d]   vf: [b][h][d][n]   ->  att[b][n][c], c=d*H+h
//    K/V tiles staged with CDNA5 async global->LDS (ASYNCcnt).
// =====================================================================
__global__ __launch_bounds__(128) void attn_kernel(
    const _Float16* __restrict__ qf, const _Float16* __restrict__ kf,
    const _Float16* __restrict__ vf, const int* __restrict__ key_mask,
    _Float16* __restrict__ att) {
  const int n0t = blockIdx.x;  // 32 query tiles
  const int h   = blockIdx.y;
  const int b   = blockIdx.z;
  const int tid = threadIdx.x;
  const int lid = tid & 31, w = tid >> 5;
  const int lo = lid & 15, hi = lid >> 4;

  __shared__ __align__(16) _Float16 Qt[64][72];       // [qrow][d]
  __shared__ __align__(16) _Float16 Kt[64][72];       // [key][d] (k-major for S)
  __shared__ __align__(16) _Float16 VtT[64][72];      // [d][key] (k-major for PV)
  __shared__ __align__(16) _Float16 Pt[4][16][72];    // per-wave [m][key]

  const size_t hb = ((size_t)b * H_SZ + h) * N_SZ;       // rows of q/k
  const size_t vb = ((size_t)b * H_SZ + h) * HD * N_SZ;  // base of v (d-major)

  // stage Q tile (b128), build per-wave A fragments
#pragma unroll
  for (int it = 0; it < 4; ++it) {
    int s = it * 128 + tid;
    int r = s >> 3, g = s & 7;
    *(h8*)&Qt[r][g * 8] = *(const h8*)&qf[(hb + n0t * 64 + r) * HD + g * 8];
  }
  __syncthreads();
  v16h qa0 = load_a(&Qt[w * 16 + lo][0], hi);
  v16h qa1 = load_a(&Qt[w * 16 + lo][32], hi);

  float rowmax[8], rowsum[8];
  v8f o[4];
#pragma unroll
  for (int e = 0; e < 8; ++e) { rowmax[e] = -3.0e38f; rowsum[e] = 0.f; }
#pragma unroll
  for (int j = 0; j < 4; ++j)
#pragma unroll
    for (int e = 0; e < 8; ++e) o[j][e] = 0.f;

  for (int kt = 0; kt < N_SZ / 64; ++kt) {
    __syncthreads();   // previous tile fully consumed
#pragma unroll
    for (int it = 0; it < 4; ++it) {   // async K/V tile staging, 16B/lane
      int s = it * 128 + tid;          // 0..511
      int r = s >> 3, g = s & 7;
      async_b128(&Kt[r][g * 8], &kf[(hb + kt * 64 + r) * HD + g * 8]);
      async_b128(&VtT[r][g * 8], &vf[vb + (size_t)r * N_SZ + kt * 64 + g * 8]);
    }
    if (kt + 1 < N_SZ / 64) {  // prefetch next tiles (global_prefetch_b8)
      __builtin_prefetch(&kf[(hb + (kt + 1) * 64) * HD + tid * 32], 0, 0);
      __builtin_prefetch(&vf[vb + (kt + 1) * 64 + (size_t)(tid >> 1) * N_SZ +
                             (tid & 1) * 32], 0, 0);
    }
    wait_async0();
    __syncthreads();

    // S = Q K^T : Kt[key][d] is k-major for B (k=d, n=key)
    v8f s[4];
#pragma unroll
    for (int j = 0; j < 4; ++j) {
#pragma unroll
      for (int e = 0; e < 8; ++e) s[j][e] = 0.f;
      s[j] = wmma16(qa0, load_bT(&Kt[j * 16 + lo][0], hi), s[j]);
      s[j] = wmma16(qa1, load_bT(&Kt[j * 16 + lo][32], hi), s[j]);
    }
    // scale + mask (select, as in reference)
#pragma unroll
    for (int j = 0; j < 4; ++j) {
      int km = key_mask[b * N_SZ + kt * 64 + j * 16 + lo];
#pragma unroll
      for (int e = 0; e < 8; ++e) {
        float v = s[j][e] * 0.125f;  // 1/sqrt(64), TEMP=1
        s[j][e] = (km == 0) ? -10000000.0f : v;
      }
    }
    // online softmax; rows e+8*hi live in 16-lane halves (xor 1/2/4/8)
    float scl[8];
#pragma unroll
    for (int e = 0; e < 8; ++e) {
      float m4 = fmaxf(fmaxf(s[0][e], s[1][e]), fmaxf(s[2][e], s[3][e]));
#pragma unroll
      for (int off = 1; off < 16; off <<= 1)
        m4 = fmaxf(m4, __shfl_xor(m4, off, 32));
      float mn = fmaxf(rowmax[e], m4);
      scl[e] = __expf(rowmax[e] - mn);
      rowmax[e] = mn;
    }
    float psum[8];
#pragma unroll
    for (int e = 0; e < 8; ++e) psum[e] = 0.f;
#pragma unroll
    for (int j = 0; j < 4; ++j)
#pragma unroll
      for (int e = 0; e < 8; ++e) {
        float p = __expf(s[j][e] - rowmax[e]);
        psum[e] += p;
        Pt[w][e + 8 * hi][j * 16 + lo] = (_Float16)p;
      }
#pragma unroll
    for (int e = 0; e < 8; ++e) {
      float ps = psum[e];
#pragma unroll
      for (int off = 1; off < 16; off <<= 1) ps += __shfl_xor(ps, off, 32);
      rowsum[e] = rowsum[e] * scl[e] + ps;
    }
#pragma unroll
    for (int j = 0; j < 4; ++j)
#pragma unroll
      for (int e = 0; e < 8; ++e) o[j][e] *= scl[e];

    __syncthreads();  // Pt visible across lanes

    // O += P V : A = Pt rows (contiguous keys), B = VtT rows (k-major keys)
    v16h pa0 = load_a(&Pt[w][lo][0], hi);
    v16h pa1 = load_a(&Pt[w][lo][32], hi);
#pragma unroll
    for (int dj = 0; dj < 4; ++dj) {
      o[dj] = wmma16(pa0, load_bT(&VtT[dj * 16 + lo][0], hi), o[dj]);
      o[dj] = wmma16(pa1, load_bT(&VtT[dj * 16 + lo][32], hi), o[dj]);
    }
  }

  // epilogue: normalize, write att[b][n][c], c = d*H + h
#pragma unroll
  for (int e = 0; e < 8; ++e) rowsum[e] = 1.0f / rowsum[e];
#pragma unroll
  for (int dj = 0; dj < 4; ++dj)
#pragma unroll
    for (int e = 0; e < 8; ++e) {
      int qrow = n0t * 64 + w * 16 + e + 8 * hi;
      int dloc = dj * 16 + lo;
      int c = dloc * H_SZ + h;
      att[((size_t)b * N_SZ + qrow) * CH + c] = (_Float16)(o[dj][e] * rowsum[e]);
    }
}

// =====================================================================
// 3) Output projection + bias + residual: y[b][n][c] fp32.
//    att[b][n][c] rows are contiguous in c => transposed staging is b128.
// =====================================================================
__global__ __launch_bounds__(128) void out_gemm_kernel(
    const float* __restrict__ Wm, const float* __restrict__ bm,
    const _Float16* __restrict__ att, const float* __restrict__ fq,
    float* __restrict__ y) {
  const int ct = blockIdx.x;
  const int nt = blockIdx.y;
  const int b  = blockIdx.z;
  const int tid = threadIdx.x;
  const int lid = tid & 31, w = tid >> 5;
  const int lo = lid & 15, hi = lid >> 4;

  __shared__ __align__(16) _Float16 Wt[64][40];    // [c_out][k] row-major
  __shared__ __align__(16) _Float16 XtT[64][40];   // [n][k] k-major

  v8f acc[4];
#pragma unroll
  for (int j = 0; j < 4; ++j)
#pragma unroll
    for (int e = 0; e < 8; ++e) acc[j][e] = 0.f;

  for (int kk = 0; kk < CH; kk += 32) {
    __syncthreads();
#pragma unroll
    for (int it = 0; it < 4; ++it) {   // Wt: f32x4 load -> cvt -> h4 store
      int s = it * 128 + tid;
      int r = s >> 3, g = s & 7;
      f4 x = *(const f4*)&Wm[(size_t)(ct * 64 + r) * CH + kk + g * 4];
      *(h4*)&Wt[r][g * 4] = __builtin_convertvector(x, h4);
    }
#pragma unroll
    for (int it = 0; it < 2; ++it) {   // XtT: b128 both sides
      int s = it * 128 + tid;          // 0..255
      int r = s >> 2, g = s & 3;
      *(h8*)&XtT[r][g * 8] =
          *(const h8*)&att[((size_t)b * N_SZ + nt * 64 + r) * CH + kk + g * 8];
    }
    __syncthreads();

    v16h a = load_a(&Wt[w * 16 + lo][0], hi);
    v16h b0 = load_bT(&XtT[0 * 16 + lo][0], hi);
    v16h b1 = load_bT(&XtT[1 * 16 + lo][0], hi);
    v16h b2 = load_bT(&XtT[2 * 16 + lo][0], hi);
    v16h b3 = load_bT(&XtT[3 * 16 + lo][0], hi);
    acc[0] = wmma16(a, b0, acc[0]);
    acc[1] = wmma16(a, b1, acc[1]);
    acc[2] = wmma16(a, b2, acc[2]);
    acc[3] = wmma16(a, b3, acc[3]);
  }

#pragma unroll
  for (int j = 0; j < 4; ++j)
#pragma unroll
    for (int e = 0; e < 8; ++e) {
      int c = ct * 64 + w * 16 + e + 8 * hi;
      int n = nt * 64 + j * 16 + lo;
      float v = acc[j][e] + bm[c] + fq[((size_t)b * CH + c) * N_SZ + n];
      y[((size_t)b * N_SZ + n) * CH + c] = v;
    }
}

// =====================================================================
// 4) Channel LayerNorm per (b, n); writes final out[b][c][n]
// =====================================================================
__global__ __launch_bounds__(256) void ln_kernel(
    const float* __restrict__ y, const float* __restrict__ g,
    const float* __restrict__ be, float* __restrict__ out) {
  const int n = blockIdx.x;
  const int b = blockIdx.y;
  const int tid = threadIdx.x;
  const int lid = tid & 31, w = tid >> 5;
  const float* row = y + ((size_t)b * N_SZ + n) * CH;

  float v0 = row[tid], v1 = row[tid + 256];
  float s = v0 + v1, q = v0 * v0 + v1 * v1;
#pragma unroll
  for (int off = 1; off < 32; off <<= 1) {
    s += __shfl_xor(s, off, 32);
    q += __shfl_xor(q, off, 32);
  }
  __shared__ float ws_s[8], ws_q[8];
  __shared__ float mu_s, rs_s;
  if (lid == 0) { ws_s[w] = s; ws_q[w] = q; }
  __syncthreads();
  if (tid == 0) {
    float S = 0.f, Q = 0.f;
#pragma unroll
    for (int i = 0; i < 8; ++i) { S += ws_s[i]; Q += ws_q[i]; }
    float mu = S * (1.0f / CH);
    float var = Q * (1.0f / CH) - mu * mu;
    mu_s = mu;
    rs_s = rsqrtf(var + 1e-5f);
  }
  __syncthreads();
  float mu = mu_s, rs = rs_s;
  out[((size_t)b * CH + tid) * N_SZ + n] = (v0 - mu) * rs * g[tid] + be[tid];
  out[((size_t)b * CH + tid + 256) * N_SZ + n] =
      (v1 - mu) * rs * g[tid + 256] + be[tid + 256];
}

extern "C" void kernel_launch(void* const* d_in, const int* in_sizes, int n_in,
                              void* d_out, int out_size, void* d_ws,
                              size_t ws_size, hipStream_t stream) {
  (void)in_sizes; (void)n_in; (void)out_size; (void)ws_size;
  const float* fq  = (const float*)d_in[0];
  const float* fk  = (const float*)d_in[1];
  const int*   km  = (const int*)d_in[2];
  const float* Wq  = (const float*)d_in[3];
  const float* bq  = (const float*)d_in[4];
  const float* Wk  = (const float*)d_in[5];
  const float* bk  = (const float*)d_in[6];
  const float* Wf  = (const float*)d_in[7];
  const float* bf  = (const float*)d_in[8];
  const float* Wm  = (const float*)d_in[9];
  const float* bm  = (const float*)d_in[10];
  const float* lng = (const float*)d_in[11];
  const float* lnb = (const float*)d_in[12];

  char* ws = (char*)d_ws;
  _Float16* qf = (_Float16*)(ws);                  // 4 MB [b][h][n][d]
  _Float16* kf = (_Float16*)(ws + (4u << 20));     // 4 MB [b][h][n][d]
  _Float16* vf = (_Float16*)(ws + (8u << 20));     // 4 MB [b][h][d][n]
  _Float16* at = (_Float16*)(ws + (12u << 20));    // 4 MB [b][n][c]
  float*    yb = (float*)   (ws + (16u << 20));    // 8 MB [b][n][c]

  dim3 pg(CH / 64, N_SZ / 64, B_SZ);
  proj_kernel<<<pg, 128, 0, stream>>>(Wq, bq, fq, qf, 0);
  proj_kernel<<<pg, 128, 0, stream>>>(Wk, bk, fk, kf, 0);
  proj_kernel<<<pg, 128, 0, stream>>>(Wf, bf, fk, vf, 1);
  attn_kernel<<<dim3(N_SZ / 64, H_SZ, B_SZ), 128, 0, stream>>>(qf, kf, vf, km, at);
  out_gemm_kernel<<<pg, 128, 0, stream>>>(Wm, bm, at, fq, yb);
  ln_kernel<<<dim3(N_SZ, B_SZ), 256, 0, stream>>>(yb, lng, lnb, (float*)d_out);
}
